// DGCNN_67903432949909
// MI455X (gfx1250) — matrix operand order repaired and checked by reading, not compiled, not run on hardware
//
#include <hip/hip_runtime.h>

// ---------------------------------------------------------------------------
// DGCNN (3x DynamicEdgeConv) for gfx1250 (MI455X, wave32, WMMA)
//   B=8, N=4096, K=20, H=64, leaky_relu slope 0.2
// All matrix math via v_wmma_f32_16x16x32_f16 (f16 in, f32 accumulate).
// ---------------------------------------------------------------------------

typedef __attribute__((ext_vector_type(16))) _Float16 v16h;
typedef __attribute__((ext_vector_type(8)))  float    v8f;

#define BB     8
#define NN     4096
#define HH     64
#define KNN    20
#define FS     64          // feature stride in f32 elements (padded)
#define SLOPE  0.2f

__device__ __forceinline__ float lrelu(float x) { return x > 0.f ? x : SLOPE * x; }

// A-tile (16x32 f16): lane l -> row m=l&15, half kh=l>>4;
// holds K = [kh*16, kh*16+16) of the 32-K chunk (matches B-tile order).
__device__ __forceinline__ v16h load_a_tile_lds(const _Float16* row, int kc, int kh) {
  const _Float16* s0 = row + kc * 32 + kh * 16;
  const _Float16* s1 = row + kc * 32 + kh * 16 + 8;
  v16h a;
#pragma unroll
  for (int i = 0; i < 8; ++i) { a[i] = s0[i]; a[8 + i] = s1[i]; }
  return a;
}

// B-tile (32x16 f16): lane l -> column n=l&15, K = [kh*16, kh*16+16) of chunk.
// WT is stored transposed in LDS: WT[col][k], so the 16 K values are contiguous.
__device__ __forceinline__ v16h load_b_tile_lds(const _Float16* WT, int ke, int nc,
                                                int kc, int n, int kh) {
  const _Float16* s = WT + (size_t)(nc * 16 + n) * ke + kc * 32 + kh * 16;
  v16h b;
#pragma unroll
  for (int i = 0; i < 16; ++i) b[i] = s[i];
  return b;
}

// ------------------------------- prep kernels ------------------------------

__global__ void pad_pos_kernel(const float* __restrict__ pos, float* __restrict__ feat) {
  int t = blockIdx.x * blockDim.x + threadIdx.x;
  if (t >= BB * NN) return;
  const float* p = pos + (size_t)t * 3;
  float* f = feat + (size_t)t * FS;
  f[0] = p[0]; f[1] = p[1]; f[2] = p[2];
#pragma unroll
  for (int i = 3; i < FS; ++i) f[i] = 0.f;
}

__global__ void norms_kernel(const float* __restrict__ feat, float* __restrict__ norms) {
  int t = blockIdx.x * blockDim.x + threadIdx.x;
  if (t >= BB * NN) return;
  const float4* f = (const float4*)(feat + (size_t)t * FS);
  float s = 0.f;
#pragma unroll
  for (int i = 0; i < FS / 4; ++i) {
    float4 v = f[i];
    s += v.x * v.x + v.y * v.y + v.z * v.z + v.w * v.w;
  }
  norms[t] = s;
}

// ------------------------------- kNN kernel --------------------------------
// One wave = 16 queries. Gram tiles via WMMA; register-resident sorted top-20
// per (query, candidate-half); merge of the two halves at the end.

template <int KC>
__global__ __launch_bounds__(128) void knn_kernel(const float* __restrict__ feat,
                                                  const float* __restrict__ norms,
                                                  int* __restrict__ nbr) {
  __shared__ float dtile[4][16][17];
  __shared__ float md[4][16][2][KNN];
  __shared__ int   mi[4][16][2][KNN];

  const int wave = threadIdx.x >> 5;
  const int lane = threadIdx.x & 31;
  const int m    = lane & 15;
  const int kh   = lane >> 4;

  const int tile = blockIdx.x * 4 + wave;        // 256 tiles per batch
  const int b    = tile >> 8;
  const int q0   = (tile & 255) << 4;

  const float* fb = feat  + (size_t)b * NN * FS;
  const float* nb = norms + (size_t)b * NN;

  // query A-tiles (constant over the candidate loop)
  v16h aq[KC];
#pragma unroll
  for (int kc = 0; kc < KC; ++kc) {
    const float* src = fb + (size_t)(q0 + m) * FS + kc * 32;
    v16h a;
#pragma unroll
    for (int i = 0; i < 8; ++i) {
      a[i]     = (_Float16)src[kh * 16 + i];
      a[8 + i] = (_Float16)src[kh * 16 + 8 + i];
    }
    aq[kc] = a;
  }

  // query norms for this lane's C-layout rows
  float nq[8];
#pragma unroll
  for (int r = 0; r < 8; ++r) nq[r] = nb[q0 + r + 8 * kh];

  float ld[KNN];
  int   li[KNN];
#pragma unroll
  for (int t = 0; t < KNN; ++t) { ld[t] = 3.0e38f; li[t] = 0; }

  for (int ct = 0; ct < NN / 16; ++ct) {
    const int c0 = ct << 4;

    // B-tiles: lane's column = candidate (c0+m); K slice per kh.
    v8f c = {};
#pragma unroll
    for (int kc = 0; kc < KC; ++kc) {
      const float* s = fb + (size_t)(c0 + m) * FS + kc * 32 + kh * 16;
      v16h bt;
#pragma unroll
      for (int i = 0; i < 16; ++i) bt[i] = (_Float16)s[i];
      c = __builtin_amdgcn_wmma_f32_16x16x32_f16(false, aq[kc], false, bt,
                                                 (short)0, c, false, false);
    }

    if (ct + 1 < NN / 16)
      __builtin_prefetch(fb + (size_t)(c0 + 16 + m) * FS, 0, 1);

    // distances into LDS (C layout: lane -> col m, rows r+8*kh)
    const float ncn = nb[c0 + m];
#pragma unroll
    for (int r = 0; r < 8; ++r)
      dtile[wave][r + 8 * kh][m] = nq[r] + ncn - 2.0f * c[r];

    // selection: lane owns query row m, candidate half kh (8 candidates)
    for (int i = 0; i < 8; ++i) {
      float d = dtile[wave][m][kh * 8 + i];
      int   j = c0 + kh * 8 + i;
      if (d < ld[KNN - 1]) {
        ld[KNN - 1] = d; li[KNN - 1] = j;
#pragma unroll
        for (int t = KNN - 1; t > 0; --t) {
          if (ld[t] < ld[t - 1]) {
            float td = ld[t]; ld[t] = ld[t - 1]; ld[t - 1] = td;
            int   ti = li[t]; li[t] = li[t - 1]; li[t - 1] = ti;
          }
        }
      }
    }
  }

  // merge the two half-lists per query (same-wave LDS, in-order)
#pragma unroll
  for (int t = 0; t < KNN; ++t) { md[wave][m][kh][t] = ld[t]; mi[wave][m][kh][t] = li[t]; }

  if (kh == 0) {
    int a = 0, c2 = 0;
    int* out = nbr + ((size_t)b * NN + q0 + m) * KNN;
#pragma unroll
    for (int t = 0; t < KNN; ++t) {
      float d0 = (a  < KNN) ? md[wave][m][0][a]  : 3.4e38f;
      float d1 = (c2 < KNN) ? md[wave][m][1][c2] : 3.4e38f;
      if (d0 <= d1) { out[t] = mi[wave][m][0][a];  ++a; }
      else          { out[t] = mi[wave][m][1][c2]; ++c2; }
    }
  }
}

// ------------------------------- edge-MLP kernel ---------------------------
// One wave = 16 points. Loop over k=0..19 edges; three WMMA layers with
// leaky-ReLU; fp32 max accumulation in C-layout registers.

template <int DIN, int KC1>
__global__ __launch_bounds__(128) void mlp_kernel(const float* __restrict__ feat,
                                                  const int* __restrict__ nbr,
                                                  const float* __restrict__ w1,
                                                  const float* __restrict__ b1,
                                                  const float* __restrict__ w2,
                                                  const float* __restrict__ b2,
                                                  const float* __restrict__ w3,
                                                  const float* __restrict__ b3,
                                                  float* __restrict__ out) {
  constexpr int KE = KC1 * 32;               // padded edge-feature dim
  constexpr int DE = 2 * DIN;                // real edge-feature dim
  constexpr int EW = (KE > HH) ? KE : HH;    // staging row width

  __shared__ _Float16 WT1[HH][KE];           // W1^T, zero padded rows >= DE
  __shared__ _Float16 WT2[HH][HH];
  __shared__ _Float16 WT3[HH][HH];
  __shared__ float    BS[3][HH];
  __shared__ _Float16 EF[4][16][EW];         // per-wave staging

  const int tid = threadIdx.x;

  for (int i = tid; i < HH * KE; i += 128) {
    int cc = i / KE, k = i % KE;
    WT1[cc][k] = (k < DE) ? (_Float16)w1[(size_t)k * HH + cc] : (_Float16)0.f;
  }
  for (int i = tid; i < HH * HH; i += 128) {
    int cc = i / HH, k = i % HH;
    WT2[cc][k] = (_Float16)w2[(size_t)k * HH + cc];
    WT3[cc][k] = (_Float16)w3[(size_t)k * HH + cc];
  }
  for (int i = tid; i < HH; i += 128) {
    BS[0][i] = b1[i]; BS[1][i] = b2[i]; BS[2][i] = b3[i];
  }
  __syncthreads();

  const int wave = tid >> 5;
  const int lane = tid & 31;
  const int m    = lane & 15;
  const int kh   = lane >> 4;

  const int tile = blockIdx.x * 4 + wave;
  const int b    = tile >> 8;
  const int p0   = (tile & 255) << 4;

  const float* fb = feat + (size_t)b * NN * FS;
  const int*   nb = nbr  + ((size_t)b * NN + p0) * KNN;

  float bias1[4], bias2[4], bias3[4];
#pragma unroll
  for (int nc = 0; nc < 4; ++nc) {
    int col = nc * 16 + m;
    bias1[nc] = BS[0][col]; bias2[nc] = BS[1][col]; bias3[nc] = BS[2][col];
  }

  v8f mx[4];
#pragma unroll
  for (int nc = 0; nc < 4; ++nc) {
    v8f z;
#pragma unroll
    for (int r = 0; r < 8; ++r) z[r] = -3.0e38f;
    mx[nc] = z;
  }

  _Float16 (*ef)[EW] = EF[wave];

  for (int k = 0; k < KNN; ++k) {
    // ---- stage edge features [x_i , x_j - x_i] (f32 -> f16, zero pad) ----
    const int p = p0 + m;
    const float* xp = fb + (size_t)p * FS;
    if (kh == 0) {
#pragma unroll
      for (int i = 0; i < DIN; ++i) ef[m][i] = (_Float16)xp[i];
    } else {
      const int j = nb[(size_t)m * KNN + k];
      const float* xj = fb + (size_t)j * FS;
#pragma unroll
      for (int i = 0; i < DIN; ++i) ef[m][DIN + i] = (_Float16)(xj[i] - xp[i]);
#pragma unroll
      for (int i = DE; i < KE; ++i) ef[m][i] = (_Float16)0.f;
    }

    // ---- layer 1: (16 x DE) x (DE x 64) ----
    v8f c1[4];
#pragma unroll
    for (int nc = 0; nc < 4; ++nc) {
      v8f c;
#pragma unroll
      for (int r = 0; r < 8; ++r) c[r] = bias1[nc];
      c1[nc] = c;
    }
#pragma unroll
    for (int kc = 0; kc < KC1; ++kc) {
      v16h a = load_a_tile_lds(ef[m], kc, kh);
#pragma unroll
      for (int nc = 0; nc < 4; ++nc) {
        v16h bt = load_b_tile_lds(&WT1[0][0], KE, nc, kc, m, kh);
        c1[nc] = __builtin_amdgcn_wmma_f32_16x16x32_f16(false, a, false, bt,
                                                        (short)0, c1[nc], false, false);
      }
    }
    // lrelu -> restage as f16 activations (16 x 64), C layout -> LDS [row][col]
#pragma unroll
    for (int nc = 0; nc < 4; ++nc)
#pragma unroll
      for (int r = 0; r < 8; ++r)
        ef[r + 8 * kh][nc * 16 + m] = (_Float16)lrelu(c1[nc][r]);

    // ---- layer 2: (16 x 64) x (64 x 64) ----
    v8f c2[4];
#pragma unroll
    for (int nc = 0; nc < 4; ++nc) {
      v8f c;
#pragma unroll
      for (int r = 0; r < 8; ++r) c[r] = bias2[nc];
      c2[nc] = c;
    }
#pragma unroll
    for (int kc = 0; kc < 2; ++kc) {
      v16h a = load_a_tile_lds(ef[m], kc, kh);
#pragma unroll
      for (int nc = 0; nc < 4; ++nc) {
        v16h bt = load_b_tile_lds(&WT2[0][0], HH, nc, kc, m, kh);
        c2[nc] = __builtin_amdgcn_wmma_f32_16x16x32_f16(false, a, false, bt,
                                                        (short)0, c2[nc], false, false);
      }
    }
#pragma unroll
    for (int nc = 0; nc < 4; ++nc)
#pragma unroll
      for (int r = 0; r < 8; ++r)
        ef[r + 8 * kh][nc * 16 + m] = (_Float16)lrelu(c2[nc][r]);

    // ---- layer 3: (16 x 64) x (64 x 64), then max-aggregate ----
    v8f c3[4];
#pragma unroll
    for (int nc = 0; nc < 4; ++nc) {
      v8f c;
#pragma unroll
      for (int r = 0; r < 8; ++r) c[r] = bias3[nc];
      c3[nc] = c;
    }
#pragma unroll
    for (int kc = 0; kc < 2; ++kc) {
      v16h a = load_a_tile_lds(ef[m], kc, kh);
#pragma unroll
      for (int nc = 0; nc < 4; ++nc) {
        v16h bt = load_b_tile_lds(&WT3[0][0], HH, nc, kc, m, kh);
        c3[nc] = __builtin_amdgcn_wmma_f32_16x16x32_f16(false, a, false, bt,
                                                        (short)0, c3[nc], false, false);
      }
    }
#pragma unroll
    for (int nc = 0; nc < 4; ++nc)
#pragma unroll
      for (int r = 0; r < 8; ++r) {
        float v = lrelu(c3[nc][r]);
        mx[nc][r] = mx[nc][r] > v ? mx[nc][r] : v;
      }
  }

  // ---- store max-aggregated output (stride FS == HH) ----
  float* ob = out + ((size_t)b * NN + p0) * HH;
#pragma unroll
  for (int nc = 0; nc < 4; ++nc)
#pragma unroll
    for (int r = 0; r < 8; ++r)
      ob[(size_t)(r + 8 * kh) * HH + nc * 16 + m] = mx[nc][r];
}

// ------------------------------- launch ------------------------------------

extern "C" void kernel_launch(void* const* d_in, const int* in_sizes, int n_in,
                              void* d_out, int out_size, void* d_ws, size_t ws_size,
                              hipStream_t stream) {
  const float* pos = (const float*)d_in[0];
  const float* w11 = (const float*)d_in[1];  const float* b11 = (const float*)d_in[2];
  const float* w12 = (const float*)d_in[3];  const float* b12 = (const float*)d_in[4];
  const float* w13 = (const float*)d_in[5];  const float* b13 = (const float*)d_in[6];
  const float* w21 = (const float*)d_in[7];  const float* b21 = (const float*)d_in[8];
  const float* w22 = (const float*)d_in[9];  const float* b22 = (const float*)d_in[10];
  const float* w23 = (const float*)d_in[11]; const float* b23 = (const float*)d_in[12];
  const float* w31 = (const float*)d_in[13]; const float* b31 = (const float*)d_in[14];
  const float* w32 = (const float*)d_in[15]; const float* b32 = (const float*)d_in[16];
  const float* w33 = (const float*)d_in[17]; const float* b33 = (const float*)d_in[18];

  const size_t pts = (size_t)BB * NN;
  float* feat0 = (float*)d_ws;
  float* feat1 = feat0 + pts * FS;
  float* feat2 = feat1 + pts * FS;
  float* norms = feat2 + pts * FS;
  int*   nbr   = (int*)(norms + pts);

  const dim3 blk(128);
  const int gPts  = (int)((pts + 127) / 128);
  const int gTile = (int)(pts / 64);    // (pts/16 tiles) / 4 waves per block

  pad_pos_kernel<<<gPts, blk, 0, stream>>>(pos, feat0);

  // ---- layer 1 (D = 3) ----
  norms_kernel<<<gPts, blk, 0, stream>>>(feat0, norms);
  knn_kernel<1><<<gTile, blk, 0, stream>>>(feat0, norms, nbr);
  mlp_kernel<3, 1><<<gTile, blk, 0, stream>>>(feat0, nbr, w11, b11, w12, b12, w13, b13, feat1);

  // ---- layer 2 (D = 64) ----
  norms_kernel<<<gPts, blk, 0, stream>>>(feat1, norms);
  knn_kernel<2><<<gTile, blk, 0, stream>>>(feat1, norms, nbr);
  mlp_kernel<64, 4><<<gTile, blk, 0, stream>>>(feat1, nbr, w21, b21, w22, b22, w23, b23, feat2);

  // ---- layer 3 (D = 64) ----
  norms_kernel<<<gPts, blk, 0, stream>>>(feat2, norms);
  knn_kernel<2><<<gTile, blk, 0, stream>>>(feat2, norms, nbr);
  mlp_kernel<64, 4><<<gTile, blk, 0, stream>>>(feat2, nbr, w31, b31, w32, b32, w33, b33,
                                               (float*)d_out);
}
